// DeformableTransformer_28458453303454
// MI455X (gfx1250) — compile-verified, matrix operand order checked by reading, not run on hardware
//
#include <hip/hip_runtime.h>
#include <hip/hip_bf16.h>
#include <math.h>

// Problem dimensions (fixed by the reference)
#define NB   2
#define NQ   300
#define CDIM 256
#define FDIM 1024
#define NH   8
#define NL   4
#define NP   4
#define NF   5
#define LIN  20197

typedef __attribute__((ext_vector_type(16))) __bf16 v16bf;
typedef __attribute__((ext_vector_type(8)))  float  v8f;

__device__ __forceinline__ uint32_t packbf2(float lo, float hi) {
  union { __bf16 h[2]; uint32_t u; } p;
  p.h[0] = (__bf16)lo;              // lowers to v_cvt_pk_bf16_f32
  p.h[1] = (__bf16)hi;
  return p.u;
}
__device__ __forceinline__ unsigned short f2bf16(float f) {
  union { __bf16 h; unsigned short s; } p;
  p.h = (__bf16)f;
  return p.s;
}
__device__ __forceinline__ float bf2f(unsigned short u) {
  return __uint_as_float(((uint32_t)u) << 16);
}

// ---------------------------------------------------------------------------
// GEMM: D(M,N) = A(M,K) @ W(N,K)^T + bias.   bf16 WMMA, f32 accumulate.
// Block = 128 threads (4 wave32), tile 32(M) x 64(N); K-step 32.
// Each wave: 2 A-fragments vs 1 shared B-fragment -> 2 WMMAs / k-step.
// mode: 0 = f32 out, 1 = f32 + ReLU, 2 = bf16 out.
// Requires: K % 32 == 0, N % 64 == 0 (true for all call sites).
// ---------------------------------------------------------------------------
__global__ __launch_bounds__(128) void gemm_bf16_wmma_kernel(
    const float* __restrict__ A, const float* __restrict__ W,
    const float* __restrict__ bias, void* __restrict__ D,
    int M, int K, int Nout, int mode)
{
  __shared__ uint32_t alds[32 * 16];      // 32 rows x 32 bf16 (packed pairs)
  const int t    = threadIdx.x;
  const int wave = t >> 5;
  const int lane = t & 31;
  const int m0   = blockIdx.x * 32;
  const int n0   = blockIdx.y * 64 + wave * 16;
  const int row  = lane & 15;
  const int kb   = (lane >> 4) << 3;      // 0 or 8 (ISA A/B fragment layout)
  const long wrow = (long)(n0 + row) * K;

  v8f acc0 = {}, acc1 = {};
  for (int k0 = 0; k0 < K; k0 += 32) {
    { // cooperatively stage A tile (32x32 f32 -> bf16) into LDS
      const int r  = t >> 2;              // 0..31
      const int c8 = (t & 3) << 3;        // 0,8,16,24
      int ar = m0 + r; if (ar >= M) ar = M - 1;   // clamp (store is guarded)
      const float* ap = A + (long)ar * K + k0 + c8;
      const float4 f0 = *(const float4*)(ap);
      const float4 f1 = *(const float4*)(ap + 4);
      uint32_t* dst = &alds[r * 16 + (c8 >> 1)];
      dst[0] = packbf2(f0.x, f0.y);
      dst[1] = packbf2(f0.z, f0.w);
      dst[2] = packbf2(f1.x, f1.y);
      dst[3] = packbf2(f1.z, f1.w);
    }
    __syncthreads();

    union { uint32_t u[8]; v16bf v; } fa0, fa1;
    { // A fragments: runs K=[kb,kb+8) and K=[kb+16,kb+24) -> 2x ds_load_b128
      const uint32_t* p0 = &alds[row * 16 + (kb >> 1)];
      fa0.u[0] = p0[0]; fa0.u[1] = p0[1]; fa0.u[2] = p0[2];  fa0.u[3] = p0[3];
      fa0.u[4] = p0[8]; fa0.u[5] = p0[9]; fa0.u[6] = p0[10]; fa0.u[7] = p0[11];
      const uint32_t* p1 = p0 + 16 * 16;  // row + 16
      fa1.u[0] = p1[0]; fa1.u[1] = p1[1]; fa1.u[2] = p1[2];  fa1.u[3] = p1[3];
      fa1.u[4] = p1[8]; fa1.u[5] = p1[9]; fa1.u[6] = p1[10]; fa1.u[7] = p1[11];
    }
    union { __bf16 h[16]; v16bf v; } fb;
    { // B fragment: row (n0+row) of W, same K pattern, from L2-resident weights
      const float* wp = W + wrow + k0 + kb;
      const float4 w0 = *(const float4*)(wp);
      const float4 w1 = *(const float4*)(wp + 4);
      const float4 w2 = *(const float4*)(wp + 16);
      const float4 w3 = *(const float4*)(wp + 20);
      fb.h[0]  = (__bf16)w0.x; fb.h[1]  = (__bf16)w0.y;
      fb.h[2]  = (__bf16)w0.z; fb.h[3]  = (__bf16)w0.w;
      fb.h[4]  = (__bf16)w1.x; fb.h[5]  = (__bf16)w1.y;
      fb.h[6]  = (__bf16)w1.z; fb.h[7]  = (__bf16)w1.w;
      fb.h[8]  = (__bf16)w2.x; fb.h[9]  = (__bf16)w2.y;
      fb.h[10] = (__bf16)w2.z; fb.h[11] = (__bf16)w2.w;
      fb.h[12] = (__bf16)w3.x; fb.h[13] = (__bf16)w3.y;
      fb.h[14] = (__bf16)w3.z; fb.h[15] = (__bf16)w3.w;
    }
    acc0 = __builtin_amdgcn_wmma_f32_16x16x32_bf16(false, fa0.v, false, fb.v,
                                                   (short)0, acc0, false, false);
    acc1 = __builtin_amdgcn_wmma_f32_16x16x32_bf16(false, fa1.v, false, fb.v,
                                                   (short)0, acc1, false, false);
    __syncthreads();
  }

  const int col = n0 + row;           // D: lane%16 -> N, VGPR r -> M row
  const float bv = bias ? bias[col] : 0.0f;
#pragma unroll
  for (int r = 0; r < 8; ++r) {
    const int m = m0 + r + ((lane >> 4) << 3);
    if (m < M) {
      float v = acc0[r] + bv;
      if (mode == 1) v = fmaxf(v, 0.0f);
      if (mode == 2)
        ((unsigned short*)D)[(long)m * Nout + col] = f2bf16(v);
      else
        ((float*)D)[(long)m * Nout + col] = v;
    }
    const int m2 = m + 16;
    if (m2 < M) {
      float v = acc1[r] + bv;
      if (mode == 1) v = fmaxf(v, 0.0f);
      if (mode == 2)
        ((unsigned short*)D)[(long)m2 * Nout + col] = f2bf16(v);
      else
        ((float*)D)[(long)m2 * Nout + col] = v;
    }
  }
}

// ---------------------------------------------------------------------------
// Elementwise add (q + query_pos)
// ---------------------------------------------------------------------------
__global__ void add_kernel(const float* __restrict__ a,
                           const float* __restrict__ b,
                           float* __restrict__ o, int n)
{
  int i = blockIdx.x * blockDim.x + threadIdx.x;
  if (i < n) o[i] = a[i] + b[i];
}

// ---------------------------------------------------------------------------
// Softmax attention over 300 keys per (b,h,q). dh=32. Block=128 threads.
// ---------------------------------------------------------------------------
__global__ __launch_bounds__(128) void attn_kernel(
    const float* __restrict__ Q, const float* __restrict__ Kp,
    const float* __restrict__ V, float* __restrict__ Out)
{
  __shared__ float qv[32];
  __shared__ float sc[NQ];
  __shared__ float red[128];
  const int t = threadIdx.x;
  const int q = blockIdx.x, h = blockIdx.y, b = blockIdx.z;
  const long rowq = ((long)b * NQ + q) * CDIM + h * 32;
  if (t < 32) qv[t] = Q[rowq + t];
  __syncthreads();

  const float scale = 0.17677669529663687f;   // 1/sqrt(32)
  float lmax = -1e30f;
  for (int j = t; j < NQ; j += 128) {
    const float* kp = Kp + ((long)b * NQ + j) * CDIM + h * 32;
    float d = 0.0f;
#pragma unroll
    for (int e = 0; e < 32; ++e) d += qv[e] * kp[e];
    d *= scale;
    sc[j] = d;
    lmax = fmaxf(lmax, d);
  }
  red[t] = lmax; __syncthreads();
  for (int s = 64; s > 0; s >>= 1) { if (t < s) red[t] = fmaxf(red[t], red[t + s]); __syncthreads(); }
  const float mx = red[0]; __syncthreads();

  float lsum = 0.0f;
  for (int j = t; j < NQ; j += 128) { float e = __expf(sc[j] - mx); sc[j] = e; lsum += e; }
  red[t] = lsum; __syncthreads();
  for (int s = 64; s > 0; s >>= 1) { if (t < s) red[t] += red[t + s]; __syncthreads(); }
  const float inv = 1.0f / red[0]; __syncthreads();

  const int d = t & 31, chunk = t >> 5;
  float acc = 0.0f;
  for (int j = chunk; j < NQ; j += 4)
    acc += sc[j] * V[((long)b * NQ + j) * CDIM + h * 32 + d];
  red[t] = acc; __syncthreads();
  if (t < 32)
    Out[rowq + t] = (red[t] + red[t + 32] + red[t + 64] + red[t + 96]) * inv;
}

// ---------------------------------------------------------------------------
// Fused residual + LayerNorm over C=256. One row per block (256 threads).
// bcast=1: A is (N,NQ,C) broadcast over NF against B=(N,NF,NQ,C).
// ---------------------------------------------------------------------------
__global__ __launch_bounds__(256) void resln_kernel(
    const float* __restrict__ A, const float* __restrict__ Bv,
    const float* __restrict__ g, const float* __restrict__ beta,
    float* __restrict__ Out, int bcast)
{
  __shared__ float red[256];
  const int row = blockIdx.x, t = threadIdx.x;
  long arow = row;
  if (bcast) arow = (long)(row / (NF * NQ)) * NQ + (row % NQ);
  const float x = A[arow * CDIM + t] + Bv[(long)row * CDIM + t];
  red[t] = x; __syncthreads();
  for (int s = 128; s > 0; s >>= 1) { if (t < s) red[t] += red[t + s]; __syncthreads(); }
  const float mean = red[0] * (1.0f / CDIM); __syncthreads();
  const float c = x - mean;
  red[t] = c * c; __syncthreads();
  for (int s = 128; s > 0; s >>= 1) { if (t < s) red[t] += red[t + s]; __syncthreads(); }
  const float var = red[0] * (1.0f / CDIM); __syncthreads();
  Out[(long)row * CDIM + t] = c * rsqrtf(var + 1e-5f) * g[t] + beta[t];
}

// ---------------------------------------------------------------------------
// loc[n,f,q,h,l,p,c] = ref_xy + off/P * ref_wh * 0.5.  256 threads = (h,l,p,c)
// ---------------------------------------------------------------------------
__global__ __launch_bounds__(256) void loc_kernel(
    const float* __restrict__ off, const float* __restrict__ ref,
    float* __restrict__ loc_out)
{
  const int bid = blockIdx.x;         // (n, f, q)
  const int t = threadIdx.x;          // ((h*L+l)*P+p)*2+c
  const int n = bid / (NF * NQ);
  const int q = bid % NQ;
  const int c = t & 1;
  const int l = (t >> 3) & 3;
  const float ov = off[((long)n * NQ + q) * 256 + t];
  const float* rp = ref + ((long)bid * NL + l) * 4;
  loc_out[(long)bid * 256 + t] = rp[c] + ov * 0.25f * rp[2 + c] * 0.5f;
}

// ---------------------------------------------------------------------------
// aw softmax over L*P=16 within each head. Block = (n,q), 128 threads.
// ---------------------------------------------------------------------------
__global__ __launch_bounds__(128) void awsoftmax_kernel(
    const float* __restrict__ awraw, float* __restrict__ aw_out)
{
  __shared__ float b1[128], b2[128];
  const int row = blockIdx.x;          // n*NQ+q
  const int t = threadIdx.x;           // h*16 + (l*P+p)
  const int h = t >> 4;
  const float v = awraw[(long)row * 128 + t];
  b1[t] = v; __syncthreads();
  float m = -1e30f;
  for (int j = 0; j < 16; ++j) m = fmaxf(m, b1[h * 16 + j]);
  const float e = __expf(v - m);
  b2[t] = e; __syncthreads();
  float s = 0.0f;
  for (int j = 0; j < 16; ++j) s += b2[h * 16 + j];
  aw_out[(long)row * 128 + t] = e / s;
}

// ---------------------------------------------------------------------------
// Bilinear sampling + attention-weighted sum. Block = (n,f,q); 256 thr=(h,d).
// value is bf16 (N*NF, LIN, H, dh) -> halves the gather traffic.
// ---------------------------------------------------------------------------
__global__ __launch_bounds__(256) void sample_kernel(
    const unsigned short* __restrict__ value, const float* __restrict__ loc,
    const float* __restrict__ aw, const int* __restrict__ shapes,
    const int* __restrict__ lsi, float* __restrict__ t2pre)
{
  const int bid = blockIdx.x;                 // (n,f,q)
  const int t = threadIdx.x;
  const int h = t >> 5, d = t & 31;
  const int n = bid / (NF * NQ);
  const int q = bid % NQ;
  const int b = bid / NQ;                     // n*NF+f
  const long vbase = (long)b * LIN * CDIM + h * 32 + d;
  float acc = 0.0f;
  for (int l = 0; l < NL; ++l) {
    const int Hl = shapes[2 * l], Wl = shapes[2 * l + 1];
    const int start = lsi[l];
    for (int p = 0; p < NP; ++p) {
      const long lbase = (long)bid * 256 + ((h * NL + l) * NP + p) * 2;
      const float lx = loc[lbase], ly = loc[lbase + 1];
      const float w = aw[(((long)n * NQ + q) * NH + h) * 16 + l * NP + p];
      const float x = lx * Wl - 0.5f, y = ly * Hl - 0.5f;
      const float xf = floorf(x), yf = floorf(y);
      const int x0 = (int)xf, y0 = (int)yf;
      const float fx = x - xf, fy = y - yf;
      float s = 0.0f;
      if (y0 >= 0 && y0 < Hl) {
        if (x0 >= 0 && x0 < Wl)
          s += (1 - fx) * (1 - fy) * bf2f(value[vbase + (long)(start + y0 * Wl + x0) * CDIM]);
        if (x0 + 1 >= 0 && x0 + 1 < Wl)
          s += fx * (1 - fy) * bf2f(value[vbase + (long)(start + y0 * Wl + x0 + 1) * CDIM]);
      }
      if (y0 + 1 >= 0 && y0 + 1 < Hl) {
        if (x0 >= 0 && x0 < Wl)
          s += (1 - fx) * fy * bf2f(value[vbase + (long)(start + (y0 + 1) * Wl + x0) * CDIM]);
        if (x0 + 1 >= 0 && x0 + 1 < Wl)
          s += fx * fy * bf2f(value[vbase + (long)(start + (y0 + 1) * Wl + x0 + 1) * CDIM]);
      }
      acc += w * s;
    }
  }
  t2pre[(long)bid * CDIM + t] = acc;
}

// ---------------------------------------------------------------------------
// Temporal weights: tw = softmax_f(tgt_box @ tw_w^T + tw_b); t2agg = sum_f tw*t2
// Block = (n,q), 256 threads = channel.
// ---------------------------------------------------------------------------
__global__ __launch_bounds__(256) void temporal_kernel(
    const float* __restrict__ tbox, const float* __restrict__ t2,
    const float* __restrict__ tww, const float* __restrict__ twb,
    float* __restrict__ t2agg)
{
  __shared__ float red[256];
  __shared__ float wf[NF];
  const int nq = blockIdx.x;          // n*NQ+q
  const int t = threadIdx.x;
  const int n = nq / NQ, q = nq % NQ;
  const float wv = tww[t];
  float lg[NF];
  for (int f = 0; f < NF; ++f) {
    const float x = tbox[(((long)n * NF + f) * NQ + q) * CDIM + t] * wv;
    red[t] = x; __syncthreads();
    for (int s = 128; s > 0; s >>= 1) { if (t < s) red[t] += red[t + s]; __syncthreads(); }
    lg[f] = red[0] + twb[0]; __syncthreads();
  }
  if (t == 0) {
    float m = lg[0];
    for (int f = 1; f < NF; ++f) m = fmaxf(m, lg[f]);
    float s = 0.0f;
    for (int f = 0; f < NF; ++f) { wf[f] = __expf(lg[f] - m); s += wf[f]; }
    for (int f = 0; f < NF; ++f) wf[f] /= s;
  }
  __syncthreads();
  float acc = 0.0f;
  for (int f = 0; f < NF; ++f)
    acc += t2[(((long)n * NF + f) * NQ + q) * CDIM + t] * wf[f];
  t2agg[(long)nq * CDIM + t] = acc;
}

// ---------------------------------------------------------------------------
extern "C" void kernel_launch(void* const* d_in, const int* in_sizes, int n_in,
                              void* d_out, int out_size, void* d_ws, size_t ws_size,
                              hipStream_t stream)
{
  (void)in_sizes; (void)n_in; (void)out_size; (void)ws_size;
  const float* target  = (const float*)d_in[0];
  const float* tgt_box = (const float*)d_in[1];
  const float* qpos    = (const float*)d_in[2];
  const float* ref     = (const float*)d_in[3];
  const float* src     = (const float*)d_in[4];
  const int*   shapes  = (const int*)d_in[5];
  const int*   lsi     = (const int*)d_in[6];
  // params flattened in _make_params insertion order
  const float* sa_in_w  = (const float*)d_in[7];
  const float* sa_in_b  = (const float*)d_in[8];
  const float* sa_out_w = (const float*)d_in[9];
  const float* sa_out_b = (const float*)d_in[10];
  const float* sab_in_w  = (const float*)d_in[11];
  const float* sab_in_b  = (const float*)d_in[12];
  const float* sab_out_w = (const float*)d_in[13];
  const float* sab_out_b = (const float*)d_in[14];
  const float* off_w = (const float*)d_in[15];
  const float* off_b = (const float*)d_in[16];
  const float* aw_w  = (const float*)d_in[17];
  const float* aw_b  = (const float*)d_in[18];
  const float* val_w = (const float*)d_in[19];
  const float* val_b = (const float*)d_in[20];
  const float* out_w = (const float*)d_in[21];
  const float* out_b = (const float*)d_in[22];
  const float* outb_w = (const float*)d_in[23];
  const float* outb_b = (const float*)d_in[24];
  const float* lin1_w  = (const float*)d_in[25];
  const float* lin1_b  = (const float*)d_in[26];
  const float* lin2_w  = (const float*)d_in[27];
  const float* lin2_b  = (const float*)d_in[28];
  const float* lin1b_w = (const float*)d_in[29];
  const float* lin1b_b = (const float*)d_in[30];
  const float* lin2b_w = (const float*)d_in[31];
  const float* lin2b_b = (const float*)d_in[32];
  const float* n1_g  = (const float*)d_in[33];
  const float* n1_b  = (const float*)d_in[34];
  const float* n1b_g = (const float*)d_in[35];
  const float* n1b_b = (const float*)d_in[36];
  const float* n2_g  = (const float*)d_in[37];
  const float* n2_b  = (const float*)d_in[38];
  const float* n2b_g = (const float*)d_in[39];
  const float* n2b_b = (const float*)d_in[40];
  const float* n3_g  = (const float*)d_in[41];
  const float* n3_b  = (const float*)d_in[42];
  const float* n3b_g = (const float*)d_in[43];
  const float* n3b_b = (const float*)d_in[44];
  const float* tw_w  = (const float*)d_in[45];
  const float* tw_b  = (const float*)d_in[46];

  // outputs: target(2,300,256) | tgt_box(2,5,300,256) | loc(2,5,300,8,4,4,2) | aw(2,1,300,8,4,4)
  float* out_target = (float*)d_out;
  float* out_tbox   = out_target + (long)NB * NQ * CDIM;
  float* out_loc    = out_tbox + (long)NB * NF * NQ * CDIM;
  float* out_aw     = out_loc + (long)NB * NF * NQ * 256;

  // workspace bump allocator (total ~140 MB, dominated by bf16 value tensor)
  char* wsp = (char*)d_ws;
  size_t woff = 0;
  auto alloc = [&](size_t bytes) -> void* {
    void* p = wsp + woff;
    woff += (bytes + 255) & ~(size_t)255;
    return p;
  };
  const int R1 = NB * NQ;           // 600
  const int RB = NB * NF * NQ;      // 3000
  const int RV = NB * NF * LIN;     // 201970
  float* q1      = (float*)alloc((size_t)R1 * CDIM * 4);
  float* qb      = (float*)alloc((size_t)R1 * CDIM * 4);
  float* qkvA    = (float*)alloc((size_t)3 * R1 * CDIM * 4);
  float* qkvB    = (float*)alloc((size_t)3 * R1 * CDIM * 4);
  float* attnA   = (float*)alloc((size_t)R1 * CDIM * 4);
  float* attnB   = (float*)alloc((size_t)R1 * CDIM * 4);
  float* projA   = (float*)alloc((size_t)R1 * CDIM * 4);
  float* projB   = (float*)alloc((size_t)R1 * CDIM * 4);
  float* target1 = (float*)alloc((size_t)R1 * CDIM * 4);
  float* tbox1   = (float*)alloc((size_t)R1 * CDIM * 4);
  float* qb2     = (float*)alloc((size_t)R1 * CDIM * 4);
  float* offbuf  = (float*)alloc((size_t)R1 * 256 * 4);
  float* awraw   = (float*)alloc((size_t)R1 * 128 * 4);
  unsigned short* value = (unsigned short*)alloc((size_t)RV * CDIM * 2);
  float* t2pre   = (float*)alloc((size_t)RB * CDIM * 4);
  float* t2buf   = (float*)alloc((size_t)RB * CDIM * 4);
  float* tb2buf  = (float*)alloc((size_t)RB * CDIM * 4);
  float* tboxbc  = (float*)alloc((size_t)RB * CDIM * 4);
  float* hidden  = (float*)alloc((size_t)RB * FDIM * 4);
  float* ffbuf   = (float*)alloc((size_t)RB * CDIM * 4);
  float* t2agg   = (float*)alloc((size_t)R1 * CDIM * 4);
  float* target2 = (float*)alloc((size_t)R1 * CDIM * 4);

  auto gemm = [&](const float* A, const float* W, const float* B, void* Dp,
                  int M, int K, int N, int mode) {
    dim3 g((M + 31) / 32, N / 64);
    gemm_bf16_wmma_kernel<<<g, 128, 0, stream>>>(A, W, B, Dp, M, K, N, mode);
  };

  const int nel = R1 * CDIM;
  add_kernel<<<(nel + 255) / 256, 256, 0, stream>>>(target, qpos, q1, nel);
  add_kernel<<<(nel + 255) / 256, 256, 0, stream>>>(tgt_box, qpos, qb, nel);

  // --- self-attention (target path): Q,K from q1; V from target ---
  gemm(q1,     sa_in_w,                 sa_in_b,            qkvA,               R1, CDIM, CDIM, 0);
  gemm(q1,     sa_in_w + CDIM * CDIM,   sa_in_b + CDIM,     qkvA + R1 * CDIM,   R1, CDIM, CDIM, 0);
  gemm(target, sa_in_w + 2*CDIM*CDIM,   sa_in_b + 2*CDIM,   qkvA + 2*R1*CDIM,   R1, CDIM, CDIM, 0);
  // --- self-attention (box path) ---
  gemm(qb,      sab_in_w,               sab_in_b,           qkvB,               R1, CDIM, CDIM, 0);
  gemm(qb,      sab_in_w + CDIM*CDIM,   sab_in_b + CDIM,    qkvB + R1 * CDIM,   R1, CDIM, CDIM, 0);
  gemm(tgt_box, sab_in_w + 2*CDIM*CDIM, sab_in_b + 2*CDIM,  qkvB + 2*R1*CDIM,   R1, CDIM, CDIM, 0);

  dim3 ag(NQ, NH, NB);
  attn_kernel<<<ag, 128, 0, stream>>>(qkvA, qkvA + R1 * CDIM, qkvA + 2 * R1 * CDIM, attnA);
  attn_kernel<<<ag, 128, 0, stream>>>(qkvB, qkvB + R1 * CDIM, qkvB + 2 * R1 * CDIM, attnB);
  gemm(attnA, sa_out_w,  sa_out_b,  projA, R1, CDIM, CDIM, 0);
  gemm(attnB, sab_out_w, sab_out_b, projB, R1, CDIM, CDIM, 0);
  resln_kernel<<<R1, 256, 0, stream>>>(target,  projA, n2_g,  n2_b,  target1, 0);
  resln_kernel<<<R1, 256, 0, stream>>>(tgt_box, projB, n2b_g, n2b_b, tbox1,   0);

  // --- deformable cross attention ---
  add_kernel<<<(nel + 255) / 256, 256, 0, stream>>>(tbox1, qpos, qb2, nel);
  gemm(qb2, off_w, off_b, offbuf, R1, CDIM, 256, 0);
  gemm(qb2, aw_w,  aw_b,  awraw,  R1, CDIM, 128, 0);
  loc_kernel<<<RB, 256, 0, stream>>>(offbuf, ref, out_loc);
  awsoftmax_kernel<<<R1, 128, 0, stream>>>(awraw, out_aw);
  gemm(src, val_w, val_b, value, RV, CDIM, CDIM, 2);   // big GEMM, bf16 output
  sample_kernel<<<RB, 256, 0, stream>>>(value, out_loc, out_aw, shapes, lsi, t2pre);
  gemm(t2pre, out_w,  out_b,  t2buf,  RB, CDIM, CDIM, 0);
  gemm(t2pre, outb_w, outb_b, tb2buf, RB, CDIM, CDIM, 0);

  // --- box path: norm1_box + FFN_box + norm3_box -> final tgt_box ---
  resln_kernel<<<RB, 256, 0, stream>>>(tbox1, tb2buf, n1b_g, n1b_b, tboxbc, 1);
  gemm(tboxbc, lin1b_w, lin1b_b, hidden, RB, CDIM, FDIM, 1);     // ReLU fused
  gemm(hidden, lin2b_w, lin2b_b, ffbuf,  RB, FDIM, CDIM, 0);
  resln_kernel<<<RB, 256, 0, stream>>>(tboxbc, ffbuf, n3b_g, n3b_b, out_tbox, 0);

  // --- temporal aggregation + target FFN -> final target ---
  temporal_kernel<<<R1, 256, 0, stream>>>(out_tbox, t2buf, tw_w, tw_b, t2agg);
  resln_kernel<<<R1, 256, 0, stream>>>(target1, t2agg, n1_g, n1_b, target2, 0);
  gemm(target2, lin1_w, lin1_b, hidden, R1, CDIM, FDIM, 1);      // ReLU fused
  gemm(hidden,  lin2_w, lin2_b, ffbuf,  R1, FDIM, CDIM, 0);
  resln_kernel<<<R1, 256, 0, stream>>>(target2, ffbuf, n3_g, n3_b, out_target, 0);
}
